// TTransformer_86543591014956
// MI455X (gfx1250) — compile-verified, matrix operand order checked by reading, not run on hardware
//
#include <hip/hip_runtime.h>

// ---------------------------------------------------------------------------
// CDNA5 (gfx1250) fused transformer layer.
// One workgroup (256 threads = 8 wave32) handles one (b,t) sequence:
// 64 tokens x 256 channels, entirely in LDS, all GEMMs via
// v_wmma_f32_16x16x32_bf16 (bf16 operands, fp32 accumulate).
// ---------------------------------------------------------------------------

typedef __attribute__((ext_vector_type(16))) __bf16 v16bf;
typedef __attribute__((ext_vector_type(8)))  __bf16 v8bf;
typedef __attribute__((ext_vector_type(8)))  float  v8f;

#define MAT_ELEMS (256 * 256)   // one weight matrix

__device__ __forceinline__ v8f wmma_bf16(v16bf a, v16bf b, v8f c) {
  // (neg_a, A, neg_b, B, c_mod, C, reuse_a, reuse_b)
  return __builtin_amdgcn_wmma_f32_16x16x32_bf16(false, a, false, b,
                                                 (short)0, c, false, false);
}

union FragU { v16bf v; v8bf h[2]; };

// A/B operand fragment (16 rows/cols x 32 K) from a row-major bf16 matrix.
// Per ISA layout: lane holds row (lane&15); lane>>4 selects the +8 K offset;
// elements 0..7 -> K = k0 + h*8 + i, elements 8..15 -> K = k0 + 16 + h*8 + i.
__device__ __forceinline__ v16bf frag_rm(const __bf16* p, int stride, int row,
                                         int k0, int lane) {
  int hh = lane >> 4;
  const __bf16* r = p + row * stride + k0 + hh * 8;
  FragU f;
  f.h[0] = *(const v8bf*)(r);
  f.h[1] = *(const v8bf*)(r + 16);
  return f.v;
}

// Pre-swizzled weight fragment: 16 contiguous bf16 per lane (32B load).
__device__ __forceinline__ v16bf frag_w(const __bf16* ws, int nt, int kt,
                                        int lane) {
  return *(const v16bf*)(ws + ((((nt << 3) + kt) << 5) + lane) * 16);
}

// D tile (16x16 f32): VGPR r holds row r (lanes 0-15) / row r+8 (lanes 16-31),
// column = lane&15.
__device__ __forceinline__ void st_tile_bf16(__bf16* p, int stride, int row0,
                                             int col0, v8f d, int lane) {
  int c = col0 + (lane & 15);
  int r0 = row0 + ((lane >> 4) << 3);
#pragma unroll
  for (int r = 0; r < 8; ++r) p[(r0 + r) * stride + c] = (__bf16)d[r];
}

__device__ __forceinline__ void st_tile_bf16_T(__bf16* p, int stride, int row0,
                                               int col0, v8f d, int lane) {
  int c = col0 + (lane & 15);
  int r0 = row0 + ((lane >> 4) << 3);
#pragma unroll
  for (int r = 0; r < 8; ++r) p[c * stride + (r0 + r)] = (__bf16)d[r];
}

// 64x256 @ 256x256^T GEMM: wave w owns column tiles {2w, 2w+1}, all 4 row
// tiles. 8 K-tiles of 32 -> 64 v_wmma per wave.
__device__ __forceinline__ void gemm64x256(const __bf16* A, const __bf16* W,
                                           int lane, int wave, v8f acc[4][2]) {
  int m15 = lane & 15;
  for (int kt = 0; kt < 8; ++kt) {
    v16bf b0 = frag_w(W, wave * 2 + 0, kt, lane);
    v16bf b1 = frag_w(W, wave * 2 + 1, kt, lane);
#pragma unroll
    for (int mt = 0; mt < 4; ++mt) {
      v16bf a = frag_rm(A, 256, mt * 16 + m15, kt * 32, lane);
      acc[mt][0] = wmma_bf16(a, b0, acc[mt][0]);
      acc[mt][1] = wmma_bf16(a, b1, acc[mt][1]);
    }
  }
}

__device__ __forceinline__ void load_act_bf16(const float* __restrict__ g,
                                              __bf16* dst, int tid) {
  const float4* s = (const float4*)g;
  for (int i = tid; i < 4096; i += 256) {
    float4 f = s[i];
    __bf16* d = dst + i * 4;
    d[0] = (__bf16)f.x; d[1] = (__bf16)f.y;
    d[2] = (__bf16)f.z; d[3] = (__bf16)f.w;
  }
}

// ---------------------------------------------------------------------------
// Prep kernel: convert 6 fp32 weight matrices into bf16 WMMA-B-fragment order.
// Flat layout per matrix: (((nt*8 + kt)*32 + lane)*16 + e).
// ---------------------------------------------------------------------------
__global__ void swizzle_weights(const float* __restrict__ Wq,
                                const float* __restrict__ Wk,
                                const float* __restrict__ Wv,
                                const float* __restrict__ Wo,
                                const float* __restrict__ W1,
                                const float* __restrict__ W2,
                                __bf16* __restrict__ ws) {
  int idx = blockIdx.x * 256 + threadIdx.x;     // 0 .. 6*65536-1
  int mat  = idx >> 16;
  int e    = idx & 15;
  int lane = (idx >> 4) & 31;
  int kt   = (idx >> 9) & 7;
  int nt   = (idx >> 12) & 15;
  int h    = lane >> 4;
  int n = nt * 16 + (lane & 15);
  int k = kt * 32 + ((e & 8) ? 16 : 0) + h * 8 + (e & 7);
  const float* W = (mat == 0) ? Wq : (mat == 1) ? Wk : (mat == 2) ? Wv
                 : (mat == 3) ? Wo : (mat == 4) ? W1 : W2;
  ws[idx] = (__bf16)W[n * 256 + k];
}

// ---------------------------------------------------------------------------
// Main fused kernel. LDS plan (160 KB, hand-overlaid):
//   [0,32K)    sIn  : input staging, then attention context (GEMM-A for Wo)
//   [32K,64K)  sQ   : Q, later post-LN1 x (bf16, GEMM-A for W1)
//   [64K,96K)  sK   : K, later hidden h1 (bf16, GEMM-A for W2)
//   [96K,160K) region D, time-overlaid:
//       attention:  sVt (32K, V transposed) | sS (16K f32 scores) | sP (8K bf16)
//       after attn: sXf (64K f32 residual / LN buffer)
// ---------------------------------------------------------------------------
__launch_bounds__(256)
__global__ void transformer_fused(const float* __restrict__ qin,
                                  const float* __restrict__ kin,
                                  const float* __restrict__ vin,
                                  const __bf16* __restrict__ wsw,
                                  const float* __restrict__ bo,
                                  const float* __restrict__ b1,
                                  const float* __restrict__ b2,
                                  const float* __restrict__ g1,
                                  const float* __restrict__ be1,
                                  const float* __restrict__ g2,
                                  const float* __restrict__ be2,
                                  float* __restrict__ out) {
  __shared__ __align__(128) char smem[163840];
  __bf16* sIn = (__bf16*)(smem);
  __bf16* sQ  = (__bf16*)(smem + 32768);
  __bf16* sK  = (__bf16*)(smem + 65536);
  __bf16* sVt = (__bf16*)(smem + 98304);    // 256 x 64, dies before sXf born
  float*  sS  = (float*) (smem + 131072);   // 64 x 64
  __bf16* sP  = (__bf16*)(smem + 147456);   // 64 x 64
  float*  sXf = (float*) (smem + 98304);    // 64 x 256, overlays sVt/sS/sP

  const int tid = threadIdx.x, lane = tid & 31, wave = tid >> 5;
  const size_t base = (size_t)blockIdx.x * (64 * 256);

  const __bf16* wQ = wsw + 0 * MAT_ELEMS;
  const __bf16* wK = wsw + 1 * MAT_ELEMS;
  const __bf16* wV = wsw + 2 * MAT_ELEMS;
  const __bf16* wO = wsw + 3 * MAT_ELEMS;
  const __bf16* w1 = wsw + 4 * MAT_ELEMS;
  const __bf16* w2 = wsw + 5 * MAT_ELEMS;

  // Warm L2 / WGP$ for the weight block (gfx1250 global_prefetch_b8).
  __builtin_prefetch(wQ + (size_t)tid * 64, 0, 1);
  __builtin_prefetch(w1 + (size_t)tid * 64, 0, 1);

  // ---- Q / K / V projections --------------------------------------------
  load_act_bf16(qin + base, sIn, tid);
  __syncthreads();
  {
    v8f acc[4][2] = {};
    gemm64x256(sIn, wQ, lane, wave, acc);
#pragma unroll
    for (int mt = 0; mt < 4; ++mt)
#pragma unroll
      for (int j = 0; j < 2; ++j)
        st_tile_bf16(sQ, 256, mt * 16, (wave * 2 + j) * 16, acc[mt][j], lane);
  }
  __syncthreads();

  load_act_bf16(kin + base, sIn, tid);
  __syncthreads();
  {
    v8f acc[4][2] = {};
    gemm64x256(sIn, wK, lane, wave, acc);
#pragma unroll
    for (int mt = 0; mt < 4; ++mt)
#pragma unroll
      for (int j = 0; j < 2; ++j)
        st_tile_bf16(sK, 256, mt * 16, (wave * 2 + j) * 16, acc[mt][j], lane);
  }
  __syncthreads();

  load_act_bf16(vin + base, sIn, tid);
  __syncthreads();
  {
    v8f acc[4][2] = {};
    gemm64x256(sIn, wV, lane, wave, acc);
#pragma unroll
    for (int mt = 0; mt < 4; ++mt)
#pragma unroll
      for (int j = 0; j < 2; ++j)   // transposed: sVt[feature][token]
        st_tile_bf16_T(sVt, 64, mt * 16, (wave * 2 + j) * 16, acc[mt][j], lane);
  }
  __syncthreads();

  // ---- Attention: 4 heads of (64x64) @ (64x64) ---------------------------
  const int m15 = lane & 15;
  for (int h0 = 0; h0 < 4; ++h0) {
    // scores S = Q_h K_h^T * 1/sqrt(64); 16 tiles, 2 per wave
    {
      v8f sa[2] = {};
      for (int kt = 0; kt < 2; ++kt) {
#pragma unroll
        for (int j = 0; j < 2; ++j) {
          int tile = wave * 2 + j, mt = tile >> 2, st = tile & 3;
          v16bf aq = frag_rm(sQ, 256, mt * 16 + m15, h0 * 64 + kt * 32, lane);
          v16bf bk = frag_rm(sK, 256, st * 16 + m15, h0 * 64 + kt * 32, lane);
          sa[j] = wmma_bf16(aq, bk, sa[j]);
        }
      }
#pragma unroll
      for (int j = 0; j < 2; ++j) {
        int tile = wave * 2 + j, mt = tile >> 2, st = tile & 3;
        int c = st * 16 + m15;
        int r0 = mt * 16 + ((lane >> 4) << 3);
#pragma unroll
        for (int r = 0; r < 8; ++r)
          sS[(r0 + r) * 64 + c] = sa[j][r] * 0.125f;
      }
    }
    __syncthreads();

    // softmax over each row of sS -> bf16 probabilities in sP
    if (tid < 64) {
      float* row = sS + tid * 64;
      float mx = row[0];
      for (int i = 1; i < 64; ++i) mx = fmaxf(mx, row[i]);
      float sum = 0.f;
      for (int i = 0; i < 64; ++i) { float e = __expf(row[i] - mx); row[i] = e; sum += e; }
      float inv = 1.f / sum;
      __bf16* pr = sP + tid * 64;
      for (int i = 0; i < 64; ++i) pr[i] = (__bf16)(row[i] * inv);
    }
    __syncthreads();

    // ctx = P @ V_h ; write into sIn (repurposed as context buffer)
    {
      v8f ca[2] = {};
      for (int kt = 0; kt < 2; ++kt) {
#pragma unroll
        for (int j = 0; j < 2; ++j) {
          int tile = wave * 2 + j, mt = tile >> 2, dt = tile & 3;
          v16bf ap = frag_rm(sP, 64, mt * 16 + m15, kt * 32, lane);
          v16bf bv = frag_rm(sVt, 64, h0 * 64 + dt * 16 + m15, kt * 32, lane);
          ca[j] = wmma_bf16(ap, bv, ca[j]);
        }
      }
#pragma unroll
      for (int j = 0; j < 2; ++j) {
        int tile = wave * 2 + j, mt = tile >> 2, dt = tile & 3;
        st_tile_bf16(sIn, 256, mt * 16, h0 * 64 + dt * 16, ca[j], lane);
      }
    }
    __syncthreads();
  }

  // ---- Output projection + bias + residual(query) -> sXf (fp32) ---------
  {
    v8f acc[4][2] = {};
    gemm64x256(sIn, wO, lane, wave, acc);
    const float* qg = qin + base;
#pragma unroll
    for (int mt = 0; mt < 4; ++mt)
#pragma unroll
      for (int j = 0; j < 2; ++j) {
        int c = (wave * 2 + j) * 16 + m15;
        float bias = bo[c];
        int r0 = mt * 16 + ((lane >> 4) << 3);
#pragma unroll
        for (int r = 0; r < 8; ++r)
          sXf[(r0 + r) * 256 + c] = acc[mt][j][r] + bias + qg[(r0 + r) * 256 + c];
      }
  }
  __syncthreads();

  // ---- LayerNorm 1: sXf -> normalized fp32 (in place) + bf16 copy in sQ --
  if (tid < 64) {
    float* row = sXf + tid * 256;
    float m = 0.f;
    for (int c = 0; c < 256; ++c) m += row[c];
    m *= (1.f / 256.f);
    float v = 0.f;
    for (int c = 0; c < 256; ++c) { float d = row[c] - m; v += d * d; }
    float rs = rsqrtf(v * (1.f / 256.f) + 1e-5f);
    __bf16* xb = sQ + tid * 256;
    for (int c = 0; c < 256; ++c) {
      float xn = (row[c] - m) * rs * g1[c] + be1[c];
      row[c] = xn;
      xb[c] = (__bf16)xn;
    }
  }
  __syncthreads();

  // ---- FFN1: relu(x @ W1^T + b1) -> bf16 in sK ---------------------------
  {
    v8f acc[4][2] = {};
    gemm64x256(sQ, w1, lane, wave, acc);
#pragma unroll
    for (int mt = 0; mt < 4; ++mt)
#pragma unroll
      for (int j = 0; j < 2; ++j) {
        int c = (wave * 2 + j) * 16 + m15;
        float bias = b1[c];
        int r0 = mt * 16 + ((lane >> 4) << 3);
#pragma unroll
        for (int r = 0; r < 8; ++r)
          sK[(r0 + r) * 256 + c] = (__bf16)fmaxf(acc[mt][j][r] + bias, 0.f);
      }
  }
  __syncthreads();

  // ---- FFN2: h1 @ W2^T + b2 + x -> sXf (in place residual add) ----------
  {
    v8f acc[4][2] = {};
    gemm64x256(sK, w2, lane, wave, acc);
#pragma unroll
    for (int mt = 0; mt < 4; ++mt)
#pragma unroll
      for (int j = 0; j < 2; ++j) {
        int c = (wave * 2 + j) * 16 + m15;
        float bias = b2[c];
        int r0 = mt * 16 + ((lane >> 4) << 3);
#pragma unroll
        for (int r = 0; r < 8; ++r) {
          int o = (r0 + r) * 256 + c;
          sXf[o] = acc[mt][j][r] + bias + sXf[o];
        }
      }
  }
  __syncthreads();

  // ---- LayerNorm 2 -> global output (fp32) -------------------------------
  if (tid < 64) {
    float* row = sXf + tid * 256;
    float m = 0.f;
    for (int c = 0; c < 256; ++c) m += row[c];
    m *= (1.f / 256.f);
    float v = 0.f;
    for (int c = 0; c < 256; ++c) { float d = row[c] - m; v += d * d; }
    float rs = rsqrtf(v * (1.f / 256.f) + 1e-5f);
    float* orow = out + base + tid * 256;
    for (int c = 0; c < 256; ++c)
      orow[c] = (row[c] - m) * rs * g2[c] + be2[c];
  }
}

// ---------------------------------------------------------------------------
extern "C" void kernel_launch(void* const* d_in, const int* in_sizes, int n_in,
                              void* d_out, int out_size, void* d_ws,
                              size_t ws_size, hipStream_t stream) {
  const float* q   = (const float*)d_in[0];
  const float* k   = (const float*)d_in[1];
  const float* v   = (const float*)d_in[2];
  const float* Wq  = (const float*)d_in[3];
  const float* Wk  = (const float*)d_in[4];
  const float* Wv  = (const float*)d_in[5];
  const float* Wo  = (const float*)d_in[6];
  const float* bo  = (const float*)d_in[7];
  const float* W1  = (const float*)d_in[8];
  const float* b1  = (const float*)d_in[9];
  const float* W2  = (const float*)d_in[10];
  const float* b2  = (const float*)d_in[11];
  const float* g1  = (const float*)d_in[12];
  const float* be1 = (const float*)d_in[13];
  const float* g2  = (const float*)d_in[14];
  const float* be2 = (const float*)d_in[15];

  __bf16* wsw = (__bf16*)d_ws;   // 6 * 64K bf16 = 768 KB swizzled weights

  swizzle_weights<<<dim3(6 * 65536 / 256), dim3(256), 0, stream>>>(
      Wq, Wk, Wv, Wo, W1, W2, wsw);

  transformer_fused<<<dim3(32 * 64), dim3(256), 0, stream>>>(
      q, k, v, (const __bf16*)wsw, bo, b1, b2, g1, be1, g2, be2,
      (float*)d_out);
}
